// AdaptiveQualityGuidedSlotRouter_76948634075336
// MI455X (gfx1250) — matrix operand order repaired and checked by reading, not compiled, not run on hardware
//
#include <hip/hip_runtime.h>

// ---------------------------------------------------------------------------
// Types for CDNA5 WMMA (wave32, 16x16x32 bf16 -> f32)
// ---------------------------------------------------------------------------
typedef __bf16 bf16;
typedef bf16  bf16x16 __attribute__((ext_vector_type(16)));
typedef float floatx8 __attribute__((ext_vector_type(8)));

static constexpr int BB = 32;      // batch
static constexpr int NN = 8192;    // tokens
static constexpr int DD = 256;     // model dim
static constexpr int KK = 16;      // slots
static constexpr int NUM_ITERS = 3;
static constexpr float SCALE = 0.0625f;       // D^-0.5 = 1/16
static constexpr int TILE_N = 64;             // tokens per attn block
static constexpr int XB_LD  = 264;            // bf16 LDS row stride (pad 8)
static constexpr int ATT_LD = 72;             // attn bf16 LDS row stride

// ---------------------------------------------------------------------------
// WMMA fragment helpers (A/B: contiguous k = 16*(lane>>4) + elem; C/D per ISA)
// ---------------------------------------------------------------------------
__device__ inline floatx8 wmma_bf16(bf16x16 a, bf16x16 b, floatx8 c) {
  // v_wmma_f32_16x16x32_bf16 : D = A(16x32) * B(32x16) + C(16x16 f32)
  return __builtin_amdgcn_wmma_f32_16x16x32_bf16(
      /*neg_a=*/false, a, /*neg_b=*/false, b,
      /*c_mod=*/(short)0, c, /*reuse_a=*/false, /*reuse_b=*/false);
}

// A matrix 16x32 bf16, row-major source: src[m*ld + k]
// lanes 0-15: m=lane,  K = {0..15};  lanes 16-31: m=lane-16, K = {16..31}
__device__ inline bf16x16 load_a16x32(const bf16* src, int ld, int kbase) {
  int lane = threadIdx.x & 31;
  int m    = lane & 15;
  int koff = (lane >> 4) << 4;                 // 0 or 16
  const bf16* p = src + (size_t)m * ld + kbase + koff;
  bf16x16 a;
#pragma unroll
  for (int j = 0; j < 8; ++j) { a[j] = p[j]; a[8 + j] = p[8 + j]; }
  return a;
}

// B matrix 32x16: B[k][n] = src[(n0+n)*ld + kbase + k]   ("transposed" source,
// i.e. out = A * W^T with W row-major).  lane n = lane&15, K-half by lane>=16.
template <typename T>
__device__ inline bf16x16 load_b_nmajor(const T* src, int ld, int n0, int kbase) {
  int lane = threadIdx.x & 31;
  int n    = lane & 15;
  int koff = (lane >> 4) << 4;                 // 0 or 16
  const T* p = src + (size_t)(n0 + n) * ld + kbase + koff;
  bf16x16 b;
#pragma unroll
  for (int e = 0; e < 16; ++e) b[e] = (bf16)p[e];
  return b;
}

// B matrix 32x16: B[k][n] = src[(kbase+k)*ld + n0 + n]   (plain row-major source)
__device__ inline bf16x16 load_b_kmajor(const bf16* src, int ld, int kbase, int n0) {
  int lane = threadIdx.x & 31;
  int n    = lane & 15;
  int koff = (lane >> 4) << 4;
  bf16x16 b;
#pragma unroll
  for (int e = 0; e < 16; ++e) b[e] = src[(size_t)(kbase + koff + e) * ld + n0 + n];
  return b;
}

// C/D 16x16 f32: lane n = lane&15, rows m = r + (lane>=16 ? 8 : 0)
__device__ inline void store_c_bf16(bf16* p, int ld, floatx8 c) {
  int lane = threadIdx.x & 31;
  int n    = lane & 15;
  int moff = (lane >> 4) << 3;
#pragma unroll
  for (int r = 0; r < 8; ++r) p[(size_t)(moff + r) * ld + n] = (bf16)c[r];
}

__device__ inline float sigmoidf_(float x) { return 1.0f / (1.0f + __expf(-x)); }

// ---------------------------------------------------------------------------
// init: slots = broadcast(slot_inits), zero accumulators
// ---------------------------------------------------------------------------
__global__ __launch_bounds__(256) void init_kernel(
    const float* __restrict__ slot_inits, float* __restrict__ slots_f,
    bf16* __restrict__ slots_b, float* __restrict__ upd,
    float* __restrict__ win, float* __restrict__ tot) {
  int idx = blockIdx.x * 256 + threadIdx.x;          // [0, B*K*D)
  int kd  = idx & (KK * DD - 1);
  float v = slot_inits[kd];
  slots_f[idx] = v;
  slots_b[idx] = (bf16)v;
  upd[idx]     = 0.0f;
  if (idx < BB * KK) { win[idx] = 0.0f; tot[idx] = 0.0f; }
}

// ---------------------------------------------------------------------------
// q = slots @ Wq^T  (per batch, 16x256, K=256) in bf16
// ---------------------------------------------------------------------------
__global__ __launch_bounds__(256) void q_proj_kernel(
    const bf16* __restrict__ slots_b, const float* __restrict__ Wq,
    bf16* __restrict__ q_b) {
  int b    = blockIdx.x;
  int wave = threadIdx.x >> 5;
  const bf16* A = slots_b + (size_t)b * KK * DD;
  for (int dt = wave; dt < 16; dt += 8) {
    floatx8 acc = {};
#pragma unroll
    for (int kb = 0; kb < DD; kb += 32) {
      bf16x16 a  = load_a16x32(A, DD, kb);
      bf16x16 bb = load_b_nmajor(Wq, DD, dt * 16, kb);
      acc = wmma_bf16(a, bb, acc);
    }
    store_c_bf16(q_b + (size_t)b * KK * DD + dt * 16, DD, acc);
  }
}

// ---------------------------------------------------------------------------
// Fused: LN(inputs) -> k,v tiles (bf16 WMMA) -> logits (WMMA) -> softmax over
// slots -> partial updates += attn @ v (WMMA, f32 atomics).  Last iteration
// additionally writes attn_weights [B,N,K] and winner/total accumulators.
// ---------------------------------------------------------------------------
__global__ __launch_bounds__(256) void attn_kernel(
    const float* __restrict__ inputs, const float* __restrict__ ln_g,
    const float* __restrict__ ln_b, const float* __restrict__ Wk,
    const float* __restrict__ Wv, const bf16* __restrict__ q_b,
    float* __restrict__ upd_g, float* __restrict__ win,
    float* __restrict__ tot, float* __restrict__ out_attn, int last) {
  extern __shared__ char smem_raw[];
  float* attn_f = (float*)smem_raw;                    // [16][64]
  bf16*  xb     = (bf16*)(attn_f + KK * TILE_N);       // [64][XB_LD]
  bf16*  kt     = xb + TILE_N * XB_LD;                 // [64][XB_LD]
  bf16*  vt     = kt + TILE_N * XB_LD;                 // [64][XB_LD]
  bf16*  attn_b = vt + TILE_N * XB_LD;                 // [16][ATT_LD]

  const int b    = blockIdx.y;
  const int n0   = blockIdx.x * TILE_N;
  const int tid  = threadIdx.x;
  const int wave = tid >> 5;
  const int lane = tid & 31;

  // ---- Stage 1: LayerNorm -> bf16 LDS tile. 8 rows per wave, 8 elems/lane.
  {
    const float* gam = ln_g + lane * 8;
    const float* bet = ln_b + lane * 8;
#pragma unroll 2
    for (int r = 0; r < 8; ++r) {
      int row = wave * 8 + r;
      const float* src = inputs + ((size_t)b * NN + n0 + row) * DD + lane * 8;
      float v[8];
      float4 f0 = *(const float4*)src;
      float4 f1 = *(const float4*)(src + 4);
      v[0]=f0.x; v[1]=f0.y; v[2]=f0.z; v[3]=f0.w;
      v[4]=f1.x; v[5]=f1.y; v[6]=f1.z; v[7]=f1.w;
      float s = 0.f, ss = 0.f;
#pragma unroll
      for (int j = 0; j < 8; ++j) { s += v[j]; ss += v[j] * v[j]; }
#pragma unroll
      for (int off = 16; off >= 1; off >>= 1) {
        s  += __shfl_xor(s, off);
        ss += __shfl_xor(ss, off);
      }
      float mean = s * (1.0f / 256.0f);
      float var  = ss * (1.0f / 256.0f) - mean * mean;
      float inv  = rsqrtf(var + 1e-5f);
      bf16* dst = xb + (size_t)row * XB_LD + lane * 8;
#pragma unroll
      for (int j = 0; j < 8; ++j)
        dst[j] = (bf16)((v[j] - mean) * inv * gam[j] + bet[j]);
    }
  }
  __syncthreads();

  // ---- Stage 2: k/v projection, 64x256 each, K=256, bf16 WMMA.
  // 128 output tiles: mt(4) x dt(16) x mat(2); 16 tiles per wave.
  for (int t = wave; t < 128; t += 8) {
    int mt  = t & 3;
    int dt  = (t >> 2) & 15;
    int mat = t >> 6;
    const float* W  = mat ? Wv : Wk;
    bf16*       dst = mat ? vt : kt;
    floatx8 acc = {};
#pragma unroll
    for (int kb = 0; kb < DD; kb += 32) {
      bf16x16 a  = load_a16x32(xb + (size_t)(mt * 16) * XB_LD, XB_LD, kb);
      bf16x16 bb = load_b_nmajor(W, DD, dt * 16, kb);
      acc = wmma_bf16(a, bb, acc);
    }
    store_c_bf16(dst + (size_t)(mt * 16) * XB_LD + dt * 16, XB_LD, acc);
  }
  __syncthreads();

  // ---- Stage 3: logits[s][t] = scale * q . k, 16x64, K=256 (waves 0-3).
  if (wave < 4) {
    int nt = wave;
    floatx8 acc = {};
    const bf16* A = q_b + (size_t)b * KK * DD;
#pragma unroll
    for (int kb = 0; kb < DD; kb += 32) {
      bf16x16 a  = load_a16x32(A, DD, kb);
      bf16x16 bb = load_b_nmajor(kt + (size_t)(nt * 16) * XB_LD, XB_LD, 0, kb);
      acc = wmma_bf16(a, bb, acc);
    }
    int n    = lane & 15;
    int moff = (lane >> 4) << 3;
#pragma unroll
    for (int r = 0; r < 8; ++r)
      attn_f[(moff + r) * TILE_N + nt * 16 + n] = acc[r] * SCALE;
  }
  __syncthreads();

  // ---- Stage 4: softmax over the 16 slots, per token (threads 0-63).
  if (tid < TILE_N) {
    int t = tid;
    float l[KK];
    float mx = -3.4e38f;
    int   sbest = 0;
#pragma unroll
    for (int s = 0; s < KK; ++s) {
      l[s] = attn_f[s * TILE_N + t];
      if (l[s] > mx) { mx = l[s]; sbest = s; }
    }
    float sum = 0.f;
#pragma unroll
    for (int s = 0; s < KK; ++s) { l[s] = __expf(l[s] - mx); sum += l[s]; }
    float rinv = 1.0f / sum;
#pragma unroll
    for (int s = 0; s < KK; ++s) {
      float a = l[s] * rinv;
      attn_f[s * TILE_N + t] = a;
      attn_b[s * ATT_LD + t] = (bf16)a;
      if (last) out_attn[((size_t)b * NN + n0 + t) * KK + s] = a;
    }
    if (last) atomicAdd(win + b * KK + sbest, rinv);  // exp(0)/sum at winner
  }
  __syncthreads();

  // ---- Stage 5: per-slot total attention mass (last iter only).
  if (last && tid < KK) {
    float s = 0.f;
#pragma unroll 8
    for (int t = 0; t < TILE_N; ++t) s += attn_f[tid * TILE_N + t];
    atomicAdd(tot + b * KK + tid, s);
  }
  __syncthreads();

  // ---- Stage 6: updates[s][d] += attn(16x64) @ v(64x256), f32 atomics.
  for (int dt = wave; dt < 16; dt += 8) {
    floatx8 acc = {};
#pragma unroll
    for (int kb = 0; kb < TILE_N; kb += 32) {
      bf16x16 a  = load_a16x32(attn_b, ATT_LD, kb);
      bf16x16 bb = load_b_kmajor(vt, XB_LD, kb, dt * 16);
      acc = wmma_bf16(a, bb, acc);
    }
    int n    = lane & 15;
    int moff = (lane >> 4) << 3;
    float* base = upd_g + (size_t)b * KK * DD + dt * 16 + n;
#pragma unroll
    for (int r = 0; r < 8; ++r)
      atomicAdd(base + (size_t)(moff + r) * DD, acc[r]);
  }
}

// ---------------------------------------------------------------------------
// GRU cell + residual MLP on [16,256] slots, one block per batch.
// Also re-zeroes the updates accumulator for the next iteration.
// ---------------------------------------------------------------------------
__global__ __launch_bounds__(256) void slot_update_kernel(
    float* __restrict__ upd_g, float* __restrict__ slots_f,
    bf16* __restrict__ slots_b, const float* __restrict__ wih,
    const float* __restrict__ whh, const float* __restrict__ bih,
    const float* __restrict__ bhh, const float* __restrict__ mg,
    const float* __restrict__ mb, const float* __restrict__ w1,
    const float* __restrict__ b1, const float* __restrict__ w2,
    const float* __restrict__ b2) {
  extern __shared__ char smem_raw[];
  float* upd  = (float*)smem_raw;        // 4096
  float* hs   = upd  + 4096;             // 4096
  float* snew = hs   + 4096;             // 4096
  float* lnh  = snew + 4096;             // 4096
  float* hid  = lnh  + 4096;             // 8192
  float* red  = hid  + 8192;             // 544

  const int b   = blockIdx.x;
  const int tid = threadIdx.x;
  const size_t base = (size_t)b * KK * DD;

  for (int i = tid; i < KK * DD; i += 256) {
    upd[i] = upd_g[base + i];
    upd_g[base + i] = 0.0f;              // ready for next iteration
    hs[i] = slots_f[base + i];
  }
  __syncthreads();

  // GRU: per (slot, dim) output, 6 dot products of length 256.
  for (int o = tid; o < KK * DD; o += 256) {
    int s = o >> 8, dj = o & 255;
    const float* ur = upd + s * DD;
    const float* hr = hs + s * DD;
    const float *wr = wih + (size_t)dj * DD, *wz = wih + (size_t)(DD + dj) * DD,
                *wn = wih + (size_t)(2 * DD + dj) * DD;
    const float *vr = whh + (size_t)dj * DD, *vz = whh + (size_t)(DD + dj) * DD,
                *vn = whh + (size_t)(2 * DD + dj) * DD;
    float ir = bih[dj], iz = bih[DD + dj], in_ = bih[2 * DD + dj];
    float hrv = bhh[dj], hz = bhh[DD + dj], hn = bhh[2 * DD + dj];
#pragma unroll 4
    for (int i = 0; i < DD; ++i) {
      float u = ur[i], h = hr[i];
      ir += u * wr[i]; iz += u * wz[i]; in_ += u * wn[i];
      hrv += h * vr[i]; hz += h * vz[i]; hn += h * vn[i];
    }
    float r  = sigmoidf_(ir + hrv);
    float z  = sigmoidf_(iz + hz);
    float ng = tanhf(in_ + r * hn);
    snew[o]  = (1.0f - z) * ng + z * hr[dj];
  }
  __syncthreads();

  // LayerNorm over snew rows.
  {
    int row = tid >> 4, seg = tid & 15;
    float s = 0.f, ss = 0.f;
    const float* p = snew + row * DD + seg * 16;
#pragma unroll
    for (int j = 0; j < 16; ++j) { s += p[j]; ss += p[j] * p[j]; }
    red[tid] = s; red[256 + tid] = ss;
  }
  __syncthreads();
  if (tid < KK) {
    float s = 0.f, ss = 0.f;
#pragma unroll
    for (int i = 0; i < 16; ++i) { s += red[tid * 16 + i]; ss += red[256 + tid * 16 + i]; }
    float mean = s * (1.0f / 256.0f);
    float var  = ss * (1.0f / 256.0f) - mean * mean;
    red[512 + tid] = mean;
    red[528 + tid] = rsqrtf(var + 1e-5f);
  }
  __syncthreads();
  for (int o = tid; o < KK * DD; o += 256) {
    int s = o >> 8, dj = o & 255;
    lnh[o] = (snew[o] - red[512 + s]) * red[528 + s] * mg[dj] + mb[dj];
  }
  __syncthreads();

  // MLP hidden: relu(lnh @ w1^T + b1), [16,512]
  for (int o = tid; o < KK * 2 * DD; o += 256) {
    int s = o >> 9, hj = o & 511;
    const float* x = lnh + s * DD;
    const float* w = w1 + (size_t)hj * DD;
    float acc = b1[hj];
#pragma unroll 4
    for (int i = 0; i < DD; ++i) acc += x[i] * w[i];
    hid[o] = fmaxf(acc, 0.0f);
  }
  __syncthreads();

  // MLP out + residual -> new slots
  for (int o = tid; o < KK * DD; o += 256) {
    int s = o >> 8, dj = o & 255;
    const float* h = hid + s * 2 * DD;
    const float* w = w2 + (size_t)dj * 2 * DD;
    float acc = b2[dj];
#pragma unroll 4
    for (int i = 0; i < 2 * DD; ++i) acc += h[i] * w[i];
    float ns = snew[o] + acc;
    slots_f[base + o] = ns;
    slots_b[base + o] = (bf16)ns;
  }
}

// ---------------------------------------------------------------------------
// Quality blend + selection heads; writes slots, sel_mask, quality.
// ---------------------------------------------------------------------------
__global__ __launch_bounds__(256) void finalize_kernel(
    const float* __restrict__ slots_f, const float* __restrict__ win,
    const float* __restrict__ tot, const float* __restrict__ conf,
    const float* __restrict__ qw1, const float* __restrict__ qb1,
    const float* __restrict__ qw2, const float* __restrict__ qb2,
    const float* __restrict__ sw1, const float* __restrict__ sb1,
    const float* __restrict__ sw2, const float* __restrict__ sb2,
    float* __restrict__ out_slots, float* __restrict__ out_mask,
    float* __restrict__ out_q) {
  __shared__ float sl[KK * DD];        // 16 KB
  __shared__ float lqh[KK * 64];       // 4 KB
  __shared__ float shd[KK * 128];      // 8 KB
  const int b = blockIdx.x, tid = threadIdx.x;
  const size_t base = (size_t)b * KK * DD;

  for (int i = tid; i < KK * DD; i += 256) {
    sl[i] = slots_f[base + i];
    out_slots[base + i] = sl[i];
  }
  __syncthreads();

  for (int o = tid; o < KK * 64; o += 256) {
    int s = o >> 6, h = o & 63;
    const float* x = sl + s * DD;
    const float* w = qw1 + (size_t)h * DD;
    float acc = qb1[h];
#pragma unroll 4
    for (int i = 0; i < DD; ++i) acc += x[i] * w[i];
    lqh[o] = fmaxf(acc, 0.0f);
  }
  for (int o = tid; o < KK * 128; o += 256) {
    int s = o >> 7, h = o & 127;
    const float* x = sl + s * DD;
    const float* w = sw1 + (size_t)h * DD;
    float acc = sb1[h];
#pragma unroll 4
    for (int i = 0; i < DD; ++i) acc += x[i] * w[i];
    shd[o] = fmaxf(acc, 0.0f);
  }
  __syncthreads();

  if (tid < KK) {
    int s = tid;
    float lsum = qb2[0];
#pragma unroll 8
    for (int h = 0; h < 64; ++h) lsum += lqh[s * 64 + h] * qw2[h];
    float learned = sigmoidf_(lsum);
    float ssum = sb2[0];
#pragma unroll 8
    for (int h = 0; h < 128; ++h) ssum += shd[s * 128 + h] * sw2[h];
    float aq = win[b * KK + s] / (tot[b * KK + s] + 1e-8f);
    float quality = (0.4f * aq + 0.4f * learned) * 0.2f + conf[b * KK + s] * 0.8f;
    float mask = (sigmoidf_(ssum) > 0.5f && quality > 0.3f) ? 1.0f : 0.0f;
    out_q[b * KK + s]    = quality;
    out_mask[b * KK + s] = mask;
  }
}

// ---------------------------------------------------------------------------
// Host launcher
// ---------------------------------------------------------------------------
extern "C" void kernel_launch(void* const* d_in, const int* in_sizes, int n_in,
                              void* d_out, int out_size, void* d_ws, size_t ws_size,
                              hipStream_t stream) {
  const float* inputs     = (const float*)d_in[0];
  const float* conf       = (const float*)d_in[1];
  const float* slot_inits = (const float*)d_in[2];
  const float* ln_in_g    = (const float*)d_in[3];
  const float* ln_in_b    = (const float*)d_in[4];
  const float* Wq         = (const float*)d_in[5];
  const float* Wk         = (const float*)d_in[6];
  const float* Wv         = (const float*)d_in[7];
  const float* gru_wih    = (const float*)d_in[8];
  const float* gru_whh    = (const float*)d_in[9];
  const float* gru_bih    = (const float*)d_in[10];
  const float* gru_bhh    = (const float*)d_in[11];
  const float* mlp_ln_g   = (const float*)d_in[12];
  const float* mlp_ln_b   = (const float*)d_in[13];
  const float* mlp_w1     = (const float*)d_in[14];
  const float* mlp_b1     = (const float*)d_in[15];
  const float* mlp_w2     = (const float*)d_in[16];
  const float* mlp_b2     = (const float*)d_in[17];
  const float* sel_w1     = (const float*)d_in[18];
  const float* sel_b1     = (const float*)d_in[19];
  const float* sel_w2     = (const float*)d_in[20];
  const float* sel_b2     = (const float*)d_in[21];
  const float* qn_w1      = (const float*)d_in[22];
  const float* qn_b1      = (const float*)d_in[23];
  const float* qn_w2      = (const float*)d_in[24];
  const float* qn_b2      = (const float*)d_in[25];

  // Workspace carve (~1.6 MB)
  char* w = (char*)d_ws;
  float* slots_f = (float*)w;  w += (size_t)BB * KK * DD * sizeof(float);
  bf16*  slots_b = (bf16*)w;   w += (size_t)BB * KK * DD * sizeof(bf16);
  bf16*  q_b     = (bf16*)w;   w += (size_t)BB * KK * DD * sizeof(bf16);
  float* upd     = (float*)w;  w += (size_t)BB * KK * DD * sizeof(float);
  float* win     = (float*)w;  w += (size_t)BB * KK * sizeof(float);
  float* tot     = (float*)w;  w += (size_t)BB * KK * sizeof(float);

  // Output layout: slots | sel_mask | quality | attn_weights
  float* out       = (float*)d_out;
  float* out_slots = out;
  float* out_mask  = out + (size_t)BB * KK * DD;
  float* out_q     = out_mask + (size_t)BB * KK;
  float* out_attn  = out_q + (size_t)BB * KK;

  const int smem_attn = KK * TILE_N * 4 +
                        (3 * TILE_N * XB_LD + KK * ATT_LD) * (int)sizeof(bf16);
  const int smem_slot = (4 * 4096 + 8192 + 544) * (int)sizeof(float);

  init_kernel<<<(BB * KK * DD) / 256, 256, 0, stream>>>(
      slot_inits, slots_f, slots_b, upd, win, tot);

  for (int it = 0; it < NUM_ITERS; ++it) {
    q_proj_kernel<<<BB, 256, 0, stream>>>(slots_b, Wq, q_b);
    attn_kernel<<<dim3(NN / TILE_N, BB), 256, smem_attn, stream>>>(
        inputs, ln_in_g, ln_in_b, Wk, Wv, q_b, upd, win, tot, out_attn,
        it == NUM_ITERS - 1 ? 1 : 0);
    slot_update_kernel<<<BB, 256, smem_slot, stream>>>(
        upd, slots_f, slots_b, gru_wih, gru_whh, gru_bih, gru_bhh,
        mlp_ln_g, mlp_ln_b, mlp_w1, mlp_b1, mlp_w2, mlp_b2);
  }

  finalize_kernel<<<BB, 256, 0, stream>>>(
      slots_f, win, tot, conf, qn_w1, qn_b1, qn_w2, qn_b2,
      sel_w1, sel_b1, sel_w2, sel_b2, out_slots, out_mask, out_q);
}
